// EncoderAttention_39719857553635
// MI455X (gfx1250) — matrix operand order repaired and checked by reading, not compiled, not run on hardware
//
#include <hip/hip_runtime.h>
#include <hip/hip_bf16.h>

// ============================================================================
// EncoderAttention for MI455X (gfx1250), wave32 + WMMA + async-to-LDS.
//
// Kernel 1 (qkv_proj): X[4096,1024] @ W[1024,1024] (x3 Q/K/V), bias on Q/K,
//   f32->f16, output to workspace in [B,H,T,HS] f16 layout.
// Kernel 2 (attn): per (b, h, 32-row query block):
//   S = Qblk @ K^T (WMMA f16->f32; K/Q staged with GLOBAL_LOAD_ASYNC_TO_LDS,
//   ASYNCcnt), *8 scale (faithful to reference bug), rowwise softmax fully in
//   LDS (32x1024 f32 block resident, float4 passes), normalized P streamed
//   coalesced (b128) to d_out, then O = P @ V (V staged transposed).
//
// All WMMA B tiles sit in LDS in [N][K] orientation -> every fragment load is
// contiguous ds_load_b128 x2 per lane. Roofline: bound by the 256 MB
// probability store (~11 us @ 23.3 TB/s); the 43 GFLOP of WMMA is negligible.
// ============================================================================

typedef _Float16 f16;
typedef __attribute__((ext_vector_type(16))) _Float16 v16h;
typedef __attribute__((ext_vector_type(8)))  float    v8f;

constexpr int B_ = 4, T_ = 1024, D_ = 1024, H_ = 16, HS_ = 64;

// ---------------- WMMA fragment helpers (ISA 7.12.2 layouts) ----------------

__device__ inline v8f wmma_f16(v16h a, v16h b, v8f c) {
  return __builtin_amdgcn_wmma_f32_16x16x32_f16(
      /*neg_a=*/false, a, /*neg_b=*/false, b,
      /*c_mod=*/(short)0, c, /*reuse_a=*/false, /*reuse_b=*/false);
}

// A fragment (16x32 f16), row-major tile, stride ld. lane L: m=L%16, g=L/16.
__device__ inline v16h fragA_f16(const f16* p, int row0, int k0, int ld, int lane) {
  const int m = lane & 15, g = lane >> 4;
  const f16* r = p + (row0 + m) * ld + k0;
  v16h a;
#pragma unroll
  for (int e = 0; e < 8; ++e) a[e] = r[g * 8 + e];
#pragma unroll
  for (int e = 0; e < 8; ++e) a[8 + e] = r[16 + g * 8 + e];
  return a;
}

// A fragment with on-the-fly f32->f16 (P from score LDS).
__device__ inline v16h fragA_f32(const float* p, int row0, int k0, int ld, int lane) {
  const int m = lane & 15, g = lane >> 4;
  const float* r = p + (row0 + m) * ld + k0;
  v16h a;
#pragma unroll
  for (int e = 0; e < 8; ++e) a[e] = (f16)r[g * 8 + e];
#pragma unroll
  for (int e = 0; e < 8; ++e) a[8 + e] = (f16)r[16 + g * 8 + e];
  return a;
}

// B fragment (32x16) from an [N][K]-oriented LDS tile: two 16B chunks/lane (k=8g..8g+7, 16+8g..16+8g+7).
__device__ inline v16h fragB_nk(const f16* p, int n0, int k0, int ld, int lane) {
  const int n = lane & 15, g = lane >> 4;
  const f16* r = p + (n0 + n) * ld + k0;
  v16h b;
#pragma unroll
  for (int e = 0; e < 8; ++e) b[e] = r[g * 8 + e];
#pragma unroll
  for (int e = 0; e < 8; ++e) b[8 + e] = r[16 + g * 8 + e];
  return b;
}

// ---------------- CDNA5 async memory->LDS copy (ASYNCcnt) -------------------
// global_load_async_to_lds_b128: per-lane 16B DMA, mem_addr = SADDR + VADDR,
// lds_addr = LDS_BASE + VDST. Tracked by ASYNCcnt (s_wait_asynccnt).

__device__ inline void async_ld_b128(unsigned lds_addr, unsigned goff_bytes,
                                     const void* sbase) {
  asm volatile("global_load_async_to_lds_b128 %0, %1, %2"
               :: "v"(lds_addr), "v"(goff_bytes), "s"(sbase)
               : "memory");
}
__device__ inline void wait_async0() {
  asm volatile("s_wait_asynccnt 0x0" ::: "memory");
}

// Flat LDS-aperture address -> raw LDS byte offset (ISA 10.2: addr[31:0]).
__device__ inline unsigned lds_off(const void* p) {
  return (unsigned)(uintptr_t)p;
}

// ======================= Kernel 1: QKV projections ==========================
// grid = (H=16 [N tile == head], 64 [M tiles of B*T], 3 [q/k/v]), block 256.

__global__ __launch_bounds__(256) void qkv_proj_kernel(
    const float* __restrict__ src, const float* __restrict__ Wq,
    const float* __restrict__ Wk,  const float* __restrict__ Wv,
    const float* __restrict__ bq,  const float* __restrict__ bk,
    f16* __restrict__ ws) {
  __shared__ f16 At[64 * 72];   // A: 64 rows x 64 K, row-major (padded 72)
  __shared__ f16 BtT[64 * 72];  // B: TRANSPOSED [N=64][K=64] (padded)

  const int tid = threadIdx.x;
  const int lane = tid & 31, w = tid >> 5;
  const int h  = blockIdx.x;
  const int m0 = blockIdx.y * 64;
  const int z  = blockIdx.z;  // 0=Q, 1=K, 2=V
  const float* W    = (z == 0) ? Wq : (z == 1) ? Wk : Wv;
  const float* bias = (z == 0) ? bq : (z == 1) ? bk : nullptr;
  f16* dst = ws + (size_t)z * ((size_t)B_ * H_ * T_ * HS_);

  const int mA = w & 3;
  v8f acc0 = {}, acc1 = {};

  for (int k0 = 0; k0 < D_; k0 += 64) {
    // Stage A: 64x64 f32 of src -> f16 LDS, row-major (float4 global loads)
    for (int j = tid; j < 1024; j += 256) {
      const int row = j >> 4, q4 = (j & 15) * 4;
      const float4 vv =
          *reinterpret_cast<const float4*>(src + (size_t)(m0 + row) * D_ + k0 + q4);
      f16* d = At + row * 72 + q4;
      d[0] = (f16)vv.x; d[1] = (f16)vv.y; d[2] = (f16)vv.z; d[3] = (f16)vv.w;
    }
    // Stage B transposed: BtT[n][kk] = W[k0+kk][n]
    for (int j = tid; j < 1024; j += 256) {
      const int row = j >> 4, q4 = (j & 15) * 4;
      const float4 vv = *reinterpret_cast<const float4*>(
          W + (size_t)h * D_ * HS_ + (size_t)(k0 + row) * HS_ + q4);
      BtT[(q4 + 0) * 72 + row] = (f16)vv.x;
      BtT[(q4 + 1) * 72 + row] = (f16)vv.y;
      BtT[(q4 + 2) * 72 + row] = (f16)vv.z;
      BtT[(q4 + 3) * 72 + row] = (f16)vv.w;
    }
    __syncthreads();
#pragma unroll
    for (int kk = 0; kk < 64; kk += 32) {
      const v16h a  = fragA_f16(At, mA * 16, kk, 72, lane);
      const v16h b0 = fragB_nk(BtT, (w >> 2) * 16, kk, 72, lane);
      const v16h b1 = fragB_nk(BtT, ((w >> 2) + 2) * 16, kk, 72, lane);
      acc0 = wmma_f16(a, b0, acc0);
      acc1 = wmma_f16(a, b1, acc1);
    }
    __syncthreads();
  }

  const int g = lane >> 4, n = lane & 15;
#pragma unroll
  for (int i = 0; i < 2; ++i) {
    v8f acc;
    if (i) acc = acc1; else acc = acc0;
    const int col = ((w >> 2) + 2 * i) * 16 + n;  // hs index 0..63
    const float badd = bias ? bias[h * HS_ + col] : 0.0f;
#pragma unroll
    for (int r = 0; r < 8; ++r) {
      const int mg = m0 + mA * 16 + g * 8 + r;
      const int bb = mg >> 10, t = mg & 1023;
      dst[(((size_t)(bb * H_ + h)) * T_ + t) * HS_ + col] = (f16)(acc[r] + badd);
    }
  }
}

// ======================= Kernel 2: fused attention ==========================

constexpr int ROWS = 32;    // query rows per workgroup
constexpr int CH   = 128;   // K/V rows staged per chunk
constexpr int QLD  = 72;    // padded f16 stride for Q tile
constexpr int KLD  = 72;    // K tile [s][hs] (N-major for QK^T)
constexpr int VTLD = 136;   // V^T tile [hs][s], 16B-aligned rows
constexpr int SLD  = 1032;  // padded f32 stride for score rows

constexpr size_t SM_Q   = (size_t)ROWS * QLD * sizeof(f16);      // 4608 B
constexpr size_t SM_KV  = (size_t)CH * KLD * sizeof(f16);        // 18432 B
constexpr size_t SM_S   = (size_t)ROWS * SLD * sizeof(float);    // 132096 B
constexpr size_t SM_RED = (size_t)(256 + 32 + 32) * sizeof(float);
constexpr size_t SM_TOTAL = SM_Q + SM_KV + SM_S + SM_RED;        // ~153 KB -> 2 WG/WGP

static_assert((size_t)HS_ * VTLD * sizeof(f16) <= SM_KV, "V^T tile fits KV buffer");

__global__ __launch_bounds__(256) void attn_kernel(
    const f16* __restrict__ ws, float* __restrict__ out, float* __restrict__ wgt) {
  extern __shared__ char smem[];
  f16*   Qt     = (f16*)smem;
  f16*   KVt    = Qt + ROWS * QLD;   // phase 1: K [s][hs]; phase 4: V^T [hs][s]
  float* S      = (float*)(smem + SM_Q + SM_KV);
  float* red    = S + ROWS * SLD;
  float* rowmax = red + 256;
  float* rowinv = rowmax + 32;

  const int tid  = threadIdx.x;
  const int lane = tid & 31, w = tid >> 5;
  const int t0 = blockIdx.x * ROWS;
  const int h  = blockIdx.y;
  const int b  = blockIdx.z;

  const size_t REG = (size_t)B_ * H_ * T_ * HS_;
  const size_t bh  = ((size_t)b * H_ + h) * T_ * HS_;
  const f16* Qg = ws + bh;
  const f16* Kg = ws + REG + bh;
  const f16* Vg = ws + 2 * REG + bh;

  const unsigned ldsQ  = lds_off(Qt);
  const unsigned ldsKV = lds_off(KVt);

  // Q block: 32x64 f16 via async DMA to LDS (one b128 per thread)
  {
    const int row = tid >> 3, seg = (tid & 7) * 8;
    async_ld_b128(ldsQ + (unsigned)(row * QLD + seg) * 2,
                  (unsigned)(((t0 + row) * HS_ + seg) * 2), Qg);
  }

  // ---- Phase 1: S = (Q @ K^T) * 8, kept in LDS ----
  const int mT = w & 1;
  const int nT = w >> 1;  // second tile at nT+4
  for (int s0 = 0; s0 < T_; s0 += CH) {
    __syncthreads();  // previous chunk's compute done before KVt overwrite
    // K chunk: 128x64 f16 via async DMA (4 b128 per thread)
#pragma unroll
    for (int j = 0; j < 4; ++j) {
      const int jj = tid + j * 256;
      const int row = jj >> 3, seg = (jj & 7) * 8;
      async_ld_b128(ldsKV + (unsigned)(row * KLD + seg) * 2,
                    (unsigned)((((s0 + row) * HS_) + seg) * 2), Kg);
    }
    if (s0 + CH < T_)  // global_prefetch_b8 of next K chunk
      __builtin_prefetch(Kg + (size_t)(s0 + CH) * HS_ + tid * 32, 0, 0);
    wait_async0();
    __syncthreads();
    v8f acc0 = {}, acc1 = {};
#pragma unroll
    for (int kk = 0; kk < HS_; kk += 32) {
      const v16h a  = fragA_f16(Qt, mT * 16, kk, QLD, lane);
      const v16h b0 = fragB_nk(KVt, nT * 16, kk, KLD, lane);
      const v16h b1 = fragB_nk(KVt, (nT + 4) * 16, kk, KLD, lane);
      acc0 = wmma_f16(a, b0, acc0);
      acc1 = wmma_f16(a, b1, acc1);
    }
    const int g = lane >> 4, n = lane & 15;
#pragma unroll
    for (int r = 0; r < 8; ++r) {
      const int row = mT * 16 + g * 8 + r;
      S[row * SLD + s0 + nT * 16 + n]       = acc0[r] * 8.0f;  // faithful *sqrt(HS)
      S[row * SLD + s0 + (nT + 4) * 16 + n] = acc1[r] * 8.0f;
    }
  }
  __syncthreads();

  // ---- Phase 2: rowwise softmax in LDS (8 threads/row, float4 passes) ----
  {
    const int r = tid >> 3, p = tid & 7;
    const float4* row = reinterpret_cast<const float4*>(S + r * SLD + p * 128);
    float m = -3.4e38f;
    for (int c = 0; c < 32; ++c) {
      const float4 v = row[c];
      m = fmaxf(m, fmaxf(fmaxf(v.x, v.y), fmaxf(v.z, v.w)));
    }
    red[tid] = m;
  }
  __syncthreads();
  if (tid < 32) {
    float m = red[tid * 8];
#pragma unroll
    for (int p = 1; p < 8; ++p) m = fmaxf(m, red[tid * 8 + p]);
    rowmax[tid] = m;
  }
  __syncthreads();
  {
    const int r = tid >> 3, p = tid & 7;
    const float mx = rowmax[r];
    float4* row = reinterpret_cast<float4*>(S + r * SLD + p * 128);
    float s = 0.0f;
    for (int c = 0; c < 32; ++c) {
      float4 v = row[c];
      v.x = __expf(v.x - mx); v.y = __expf(v.y - mx);
      v.z = __expf(v.z - mx); v.w = __expf(v.w - mx);
      row[c] = v;
      s += (v.x + v.y) + (v.z + v.w);
    }
    red[tid] = s;
  }
  __syncthreads();
  if (tid < 32) {
    float s = red[tid * 8];
#pragma unroll
    for (int p = 1; p < 8; ++p) s += red[tid * 8 + p];
    rowinv[tid] = 1.0f / s;
  }
  __syncthreads();

  // ---- Phase 3: normalize + coalesced b128 store of probabilities ----
  // weights_cat layout: [B, T, H, T] -> ((b*T+t)*H + h)*T + s
  for (int i = tid; i < ROWS * T_ / 4; i += 256) {
    const int r = i >> 8, c4 = (i & 255) * 4;
    const float inv = rowinv[r];
    float4 v = *reinterpret_cast<float4*>(S + r * SLD + c4);
    v.x *= inv; v.y *= inv; v.z *= inv; v.w *= inv;
    *reinterpret_cast<float4*>(S + r * SLD + c4) = v;
    *reinterpret_cast<float4*>(
        wgt + (((size_t)(b * T_ + t0 + r)) * H_ + h) * T_ + c4) = v;
  }
  __syncthreads();

  // ---- Phase 4: O = P @ V (V staged TRANSPOSED: VtT[hs][s_local]) ----
  v8f acco = {};
  const int mO = w & 1, nO = w >> 1;
  for (int s0 = 0; s0 < T_; s0 += CH) {
    for (int j = tid; j < CH * HS_ / 8; j += 256) {
      const int row = j >> 3, seg = (j & 7) * 8;  // row = s_local, seg = hs base
      const uint4 pk =
          *reinterpret_cast<const uint4*>(Vg + (size_t)(s0 + row) * HS_ + seg);
      const f16* e8 = reinterpret_cast<const f16*>(&pk);
#pragma unroll
      for (int i = 0; i < 8; ++i) KVt[(seg + i) * VTLD + row] = e8[i];
    }
    if (s0 + CH < T_)
      __builtin_prefetch(Vg + (size_t)(s0 + CH) * HS_ + tid * 32, 0, 0);
    __syncthreads();
#pragma unroll
    for (int kc = 0; kc < CH; kc += 32) {
      const v16h a  = fragA_f32(S, mO * 16, s0 + kc, SLD, lane);
      const v16h bb = fragB_nk(KVt, nO * 16, kc, VTLD, lane);
      acco = wmma_f16(a, bb, acco);
    }
    __syncthreads();
  }

  // out layout: [B, T, H, HS] -> ((b*T+t)*H + h)*HS + e
  const int g2 = lane >> 4, n2 = lane & 15;
#pragma unroll
  for (int r = 0; r < 8; ++r) {
    const int row = mO * 16 + g2 * 8 + r;
    out[(((size_t)(b * T_ + t0 + row)) * H_ + h) * HS_ + nO * 16 + n2] = acco[r];
  }
}

// ================================ Launch ====================================

extern "C" void kernel_launch(void* const* d_in, const int* in_sizes, int n_in,
                              void* d_out, int out_size, void* d_ws, size_t ws_size,
                              hipStream_t stream) {
  const float* src = (const float*)d_in[0];
  const float* Wq  = (const float*)d_in[1];
  const float* bq  = (const float*)d_in[2];
  const float* Wk  = (const float*)d_in[3];
  const float* bk  = (const float*)d_in[4];
  const float* Wv  = (const float*)d_in[5];

  f16*   ws  = (f16*)d_ws;  // needs 3 * B*H*T*HS f16 = 24 MB of scratch
  float* out = (float*)d_out;
  float* wgt = out + (size_t)B_ * T_ * D_;

  (void)hipFuncSetAttribute((const void*)attn_kernel,
                            hipFuncAttributeMaxDynamicSharedMemorySize,
                            (int)SM_TOTAL);

  qkv_proj_kernel<<<dim3(H_, (B_ * T_) / 64, 3), dim3(256), 0, stream>>>(
      src, Wq, Wk, Wv, bq, bk, ws);
  attn_kernel<<<dim3(T_ / ROWS, H_, B_), dim3(256), SM_TOTAL, stream>>>(
      ws, out, wgt);
}
